// DualTrackBlock_11106785427544
// MI455X (gfx1250) — compile-verified, matrix operand order checked by reading, not run on hardware
//
#include <hip/hip_runtime.h>
#include <hip/hip_bf16.h>
#include <math.h>

typedef __attribute__((ext_vector_type(16))) _Float16 v16h;
typedef __attribute__((ext_vector_type(8)))  _Float16 h8;
typedef __attribute__((ext_vector_type(8)))  float    v8f;
typedef __attribute__((ext_vector_type(4)))  int      v4i;

// Problem constants (from reference setup_inputs)
constexpr int Bb   = 2;
constexpr int Lseq = 2048;
constexpr int DM   = 1024;   // d_model
constexpr int DI   = 2048;   // d_inner
constexpr int NST  = 16;     // state dim n
constexpr int Rr   = 64;     // dt rank
constexpr int XD   = Rr + 2 * NST;  // 96
constexpr int BLrows = Bb * Lseq;   // 4096

// Chunked scan config
constexpr int NCH  = 16;             // chunks per sequence
constexpr int CLEN = Lseq / NCH;     // 128 steps per chunk

// GEMM tiling: 256 threads = 8 waves (4x2); each wave owns a 32x64 output tile
// built from 2x4 16x16x32 WMMA sub-tiles. Block tile 128x128, K-step 32.
constexpr int BM = 128;
constexpr int BN = 128;
constexpr int BK = 32;
constexpr int LDS_STRIDE = 40;  // halves per row (32 + 8 pad = 80B, 16B aligned)

enum { EPI_STORE = 0, EPI_SOFTPLUS = 1, EPI_SCALE_STORE = 2, EPI_SCALE_ADD = 3 };

// ---- CDNA5 async global->LDS staging (ASYNCcnt path), guarded --------------
#if defined(__has_builtin)
#if __has_builtin(__builtin_amdgcn_global_load_async_to_lds_b128)
#define HAVE_ASYNC_LDS 1
#endif
#endif
#ifndef HAVE_ASYNC_LDS
#define HAVE_ASYNC_LDS 0
#endif

#if HAVE_ASYNC_LDS
typedef __attribute__((address_space(1))) v4i* gv4i_p;   // global int4*
typedef __attribute__((address_space(3))) v4i* lv4i_p;   // LDS int4*

__device__ __forceinline__ void async_copy16(_Float16* lds_dst,
                                             const _Float16* gsrc) {
  __builtin_amdgcn_global_load_async_to_lds_b128(
      (gv4i_p)gsrc, (lv4i_p)lds_dst, 0, 0);
}
__device__ __forceinline__ void wait_async0() {
#if defined(__has_builtin) && __has_builtin(__builtin_amdgcn_s_wait_asynccnt)
  __builtin_amdgcn_s_wait_asynccnt(0);
#else
  asm volatile("s_wait_asynccnt 0x0" ::: "memory");
#endif
}
#endif

// ---------------------------------------------------------------------------
// C[M,N] (f32) = A[M,K] (f16, lda) * W[N,K]^T (f16, ldw), fused epilogue.
// ---------------------------------------------------------------------------
__global__ __launch_bounds__(256) void gemm_wmma_kernel(
    const _Float16* __restrict__ A, int lda,
    const _Float16* __restrict__ W, int ldw,
    float* __restrict__ C, int ldc,
    int M, int N, int K,
    const float* __restrict__ bias,
    const float* __restrict__ scale_ptr,
    int mode)
{
  __shared__ alignas(16) _Float16 sA[BM * LDS_STRIDE];
  __shared__ alignas(16) _Float16 sB[BN * LDS_STRIDE];

  const int tid  = threadIdx.x;
  const int m0   = blockIdx.y * BM;
  const int n0   = blockIdx.x * BN;
  const int wave = tid >> 5;
  const int lane = tid & 31;
  const int wr   = wave >> 1;   // 0..3 : m-offset wr*32
  const int wn   = wave & 1;    // 0..1 : n-offset wn*64
  const int g    = lane >> 4;   // K-chunk group within fragment
  const int lr   = lane & 15;   // row/col within 16

  const bool bfull = (n0 + BN <= N);   // block-uniform

  const v8f vzero = {0.f, 0.f, 0.f, 0.f, 0.f, 0.f, 0.f, 0.f};
  v8f acc[2][4];
  #pragma unroll
  for (int mt = 0; mt < 2; ++mt)
    #pragma unroll
    for (int nt = 0; nt < 4; ++nt) acc[mt][nt] = vzero;

  for (int k0 = 0; k0 < K; k0 += BK) {
    // --- stage A tile (128 rows x 4 x 16B) and B tile (same): 2 chunks/thread
#if HAVE_ASYNC_LDS
    #pragma unroll
    for (int i = 0; i < 2; ++i) {
      const int chunk = i * 256 + tid;
      const int row = chunk >> 2, c = chunk & 3;
      async_copy16(&sA[row * LDS_STRIDE + c * 8],
                   A + (size_t)(m0 + row) * lda + k0 + c * 8);
    }
    if (bfull) {
      #pragma unroll
      for (int i = 0; i < 2; ++i) {
        const int chunk = i * 256 + tid;
        const int row = chunk >> 2, c = chunk & 3;
        async_copy16(&sB[row * LDS_STRIDE + c * 8],
                     W + (size_t)(n0 + row) * ldw + k0 + c * 8);
      }
    } else {
      #pragma unroll
      for (int i = 0; i < 2; ++i) {
        const int chunk = i * 256 + tid;
        const int row = chunk >> 2, c = chunk & 3;
        uint4 v = {0u, 0u, 0u, 0u};
        if (n0 + row < N)
          v = *(const uint4*)(W + (size_t)(n0 + row) * ldw + k0 + c * 8);
        *(uint4*)(&sB[row * LDS_STRIDE + c * 8]) = v;
      }
    }
    wait_async0();
#else
    #pragma unroll
    for (int i = 0; i < 2; ++i) {
      const int chunk = i * 256 + tid;
      const int row = chunk >> 2, c = chunk & 3;
      const uint4 v = *(const uint4*)(A + (size_t)(m0 + row) * lda + k0 + c * 8);
      *(uint4*)(&sA[row * LDS_STRIDE + c * 8]) = v;
    }
    #pragma unroll
    for (int i = 0; i < 2; ++i) {
      const int chunk = i * 256 + tid;
      const int row = chunk >> 2, c = chunk & 3;
      uint4 v = {0u, 0u, 0u, 0u};
      if (n0 + row < N)
        v = *(const uint4*)(W + (size_t)(n0 + row) * ldw + k0 + c * 8);
      *(uint4*)(&sB[row * LDS_STRIDE + c * 8]) = v;
    }
#endif
    __syncthreads();

    // --- load fragments (ISA 16-bit A layout: lane group g holds K chunks
    //     [8g, 8g+8) in VGPR0-3 and [16+8g, 16+8g+8) in VGPR4-7)
    v16h afrag[2], bfrag[4];
    #pragma unroll
    for (int mt = 0; mt < 2; ++mt) {
      const int row = wr * 32 + mt * 16 + lr;
      const h8 lo = *(const h8*)(&sA[row * LDS_STRIDE + g * 8]);
      const h8 hi = *(const h8*)(&sA[row * LDS_STRIDE + g * 8 + 16]);
      #pragma unroll
      for (int i = 0; i < 8; ++i) { afrag[mt][i] = lo[i]; afrag[mt][i + 8] = hi[i]; }
    }
    #pragma unroll
    for (int nt = 0; nt < 4; ++nt) {
      const int row = wn * 64 + nt * 16 + lr;
      const h8 lo = *(const h8*)(&sB[row * LDS_STRIDE + g * 8]);
      const h8 hi = *(const h8*)(&sB[row * LDS_STRIDE + g * 8 + 16]);
      #pragma unroll
      for (int i = 0; i < 8; ++i) { bfrag[nt][i] = lo[i]; bfrag[nt][i + 8] = hi[i]; }
    }

    // --- 8 WMMAs per K-step (full EXEC here; no divergence wraps these)
    #pragma unroll
    for (int mt = 0; mt < 2; ++mt)
      #pragma unroll
      for (int nt = 0; nt < 4; ++nt)
        acc[mt][nt] = __builtin_amdgcn_wmma_f32_16x16x32_f16(
            false, afrag[mt], false, bfrag[nt], (short)0, acc[mt][nt], false, false);

    __syncthreads();
  }

  const float sw = scale_ptr ? *scale_ptr : 1.0f;

  // --- epilogue + store. C/D layout: VGPR r -> M = 8*(lane/16)+r, N = lane%16
  #pragma unroll
  for (int mt = 0; mt < 2; ++mt) {
    #pragma unroll
    for (int nt = 0; nt < 4; ++nt) {
      const int n = n0 + wn * 64 + nt * 16 + lr;
      if (n >= N) continue;                       // N multiple of 16: tile-uniform
      const int mbase = m0 + wr * 32 + mt * 16 + g * 8;
      #pragma unroll
      for (int r = 0; r < 8; ++r) {
        const int m = mbase + r;
        float v = acc[mt][nt][r];
        float* cp = C + (size_t)m * ldc + n;
        if (mode == EPI_STORE) {
          if (bias) v += bias[n];
          *cp = v;
        } else if (mode == EPI_SOFTPLUS) {
          v += bias[n];
          *cp = (v > 20.f) ? v : log1pf(__expf(v));
        } else if (mode == EPI_SCALE_STORE) {
          *cp = sw * v;
        } else { // EPI_SCALE_ADD
          *cp = *cp + sw * v;
        }
      }
    }
  }
}

// ---------------------------------------------------------------------------
// LayerNorm over d_model, output f16 for the WMMA GEMMs.
// ---------------------------------------------------------------------------
__global__ __launch_bounds__(256) void layernorm_kernel(
    const float* __restrict__ x, const float* __restrict__ w,
    const float* __restrict__ b, _Float16* __restrict__ out)
{
  __shared__ float red[256];
  __shared__ float s_mean, s_rstd;
  const int row = blockIdx.x;
  const int tid = threadIdx.x;
  const float* xr = x + (size_t)row * DM;

  float s = 0.f;
  for (int i = tid; i < DM; i += 256) s += xr[i];
  red[tid] = s; __syncthreads();
  for (int o = 128; o > 0; o >>= 1) { if (tid < o) red[tid] += red[tid + o]; __syncthreads(); }
  if (tid == 0) s_mean = red[0] / DM;
  __syncthreads();
  const float m = s_mean;

  float v = 0.f;
  for (int i = tid; i < DM; i += 256) { const float d = xr[i] - m; v += d * d; }
  red[tid] = v; __syncthreads();
  for (int o = 128; o > 0; o >>= 1) { if (tid < o) red[tid] += red[tid + o]; __syncthreads(); }
  if (tid == 0) s_rstd = rsqrtf(red[0] / DM + 1e-5f);
  __syncthreads();
  const float r = s_rstd;

  for (int i = tid; i < DM; i += 256)
    out[(size_t)row * DM + i] = (_Float16)((xr[i] - m) * r * w[i] + b[i]);
}

// ---------------------------------------------------------------------------
// Depthwise causal conv (D_CONV=4) + bias + SiLU on x half of xz.
// ---------------------------------------------------------------------------
__global__ __launch_bounds__(256) void conv_silu_kernel(
    const float* __restrict__ xz, const float* __restrict__ cw,
    const float* __restrict__ cb, float* __restrict__ xs32,
    _Float16* __restrict__ xs16)
{
  const int idx = blockIdx.x * 256 + threadIdx.x;   // < B*L*DI = 8.4M
  const int ch = idx % DI;
  const int bl = idx / DI;
  const int l  = bl % Lseq;
  const float* base = xz + (size_t)bl * (2 * DI) + ch;
  float s = cb[ch];
  #pragma unroll
  for (int j = 0; j < 4; ++j) {
    const int lp = l - 3 + j;
    if (lp >= 0) s += cw[ch * 4 + j] * base[(j - 3) * (2 * DI)];
  }
  const float o = s / (1.f + __expf(-s));
  xs32[idx] = o;
  xs16[idx] = (_Float16)o;
}

// ---------------------------------------------------------------------------
// Chunked selective scan.
//   h_t = exp(dt_t * a) h_{t-1} + dt_t u_t B_t  (per channel, per state n)
// is associative across chunks: h_end = exp(a*sum(dt))*h_init + h_local.
// Phase 1: per-chunk local scan from 0, record h_local and sum(dt).
// Phase 2: tiny sequential combine over NCH chunks, 1 thread per (b,ch,n).
// Phase 3: re-scan each chunk from corrected h_init, emit y with fused
//          (+u*D) and *silu(z) epilogue.
// ---------------------------------------------------------------------------
__global__ __launch_bounds__(256) void scan_phase1_kernel(
    const float* __restrict__ dt,    // (B*L, DI)
    const float* __restrict__ u,     // (B*L, DI)
    const float* __restrict__ xdbl,  // (B*L, 96): cols [64:80] = B
    const float* __restrict__ A_log, // (DI, 16)
    float* __restrict__ hchunk,      // (B, NCH, DI, NST)
    float* __restrict__ sumdt)       // (B, NCH, DI)
{
  __shared__ float sB[2][NST];
  const int tid   = threadIdx.x;
  const int c     = blockIdx.x % NCH;
  const int chblk = (blockIdx.x / NCH) & 7;       // DI/256 = 8
  const int batch = blockIdx.x / (NCH * 8);
  const int ch    = chblk * 256 + tid;

  float a[NST], h[NST];
  #pragma unroll
  for (int n = 0; n < NST; ++n) {
    a[n] = -__expf(A_log[(size_t)ch * NST + n]);
    h[n] = 0.f;
  }
  float sdt = 0.f;

  const int t0 = c * CLEN;
  for (int t = t0; t < t0 + CLEN; ++t) {
    const size_t row = (size_t)(batch * Lseq + t);
    const int sl = t & 1;
    if (tid < NST) sB[sl][tid] = xdbl[row * XD + Rr + tid];
    __syncthreads();

    const float dtv = dt[row * DI + ch];
    const float du  = dtv * u[row * DI + ch];
    sdt += dtv;
    #pragma unroll
    for (int n = 0; n < NST; ++n)
      h[n] = __expf(dtv * a[n]) * h[n] + du * sB[sl][n];
  }
  const size_t obase = ((size_t)(batch * NCH + c) * DI + ch);
  sumdt[obase] = sdt;
  #pragma unroll
  for (int n = 0; n < NST; ++n) hchunk[obase * NST + n] = h[n];
}

__global__ __launch_bounds__(256) void scan_phase2_kernel(
    const float* __restrict__ sumdt,   // (B, NCH, DI)
    const float* __restrict__ hchunk,  // (B, NCH, DI, NST)
    const float* __restrict__ A_log,   // (DI, 16)
    float* __restrict__ hinit)         // (B, NCH, DI, NST)
{
  const int idx  = blockIdx.x * 256 + threadIdx.x;   // < B*DI*NST = 65536
  const int n    = idx % NST;
  const int ch   = (idx / NST) % DI;
  const int batch = idx / (NST * DI);
  const float a = -__expf(A_log[(size_t)ch * NST + n]);

  float h = 0.f;
  #pragma unroll
  for (int c = 0; c < NCH; ++c) {
    const size_t base = ((size_t)(batch * NCH + c) * DI + ch);
    hinit[base * NST + n] = h;
    h = __expf(a * sumdt[base]) * h + hchunk[base * NST + n];
  }
}

__global__ __launch_bounds__(256) void scan_phase3_kernel(
    const float* __restrict__ dt,    // (B*L, DI)
    const float* __restrict__ xdbl,  // (B*L, 96): B at 64, C at 80
    const float* __restrict__ u,     // (B*L, DI)
    const float* __restrict__ xz,    // (B*L, 2*DI): z at col DI+ch
    const float* __restrict__ A_log, // (DI, 16)
    const float* __restrict__ Dp,    // (DI)
    const float* __restrict__ hinit, // (B, NCH, DI, NST)
    float* __restrict__ y32,
    _Float16* __restrict__ y16)
{
  __shared__ float sB[2][NST];
  __shared__ float sC[2][NST];
  const int tid   = threadIdx.x;
  const int c     = blockIdx.x % NCH;
  const int chblk = (blockIdx.x / NCH) & 7;
  const int batch = blockIdx.x / (NCH * 8);
  const int ch    = chblk * 256 + tid;

  float a[NST], h[NST];
  const size_t ibase = ((size_t)(batch * NCH + c) * DI + ch) * NST;
  #pragma unroll
  for (int n = 0; n < NST; ++n) {
    a[n] = -__expf(A_log[(size_t)ch * NST + n]);
    h[n] = hinit[ibase + n];
  }
  const float dch = Dp[ch];

  const int t0 = c * CLEN;
  for (int t = t0; t < t0 + CLEN; ++t) {
    const size_t row = (size_t)(batch * Lseq + t);
    const int sl = t & 1;
    if (tid < NST)           sB[sl][tid]       = xdbl[row * XD + Rr + tid];
    else if (tid < 2 * NST)  sC[sl][tid - NST] = xdbl[row * XD + Rr + NST + (tid - NST)];
    __syncthreads();

    const float dtv = dt[row * DI + ch];
    const float uv  = u[row * DI + ch];
    const float du  = dtv * uv;
    float y = 0.f;
    #pragma unroll
    for (int n = 0; n < NST; ++n) {
      h[n] = __expf(dtv * a[n]) * h[n] + du * sB[sl][n];
      y += h[n] * sC[sl][n];
    }
    y += uv * dch;
    const float z = xz[row * (2 * DI) + DI + ch];
    y *= z / (1.f + __expf(-z));
    y32[row * DI + ch] = y;
    y16[row * DI + ch] = (_Float16)y;
  }
}

__global__ void cvt_f32_f16_kernel(const float* __restrict__ in,
                                   _Float16* __restrict__ out, int n)
{
  const int i = blockIdx.x * blockDim.x + threadIdx.x;
  if (i < n) out[i] = (_Float16)in[i];
}

__global__ void softmax2_kernel(const float* __restrict__ tw, float* __restrict__ w)
{
  if (threadIdx.x == 0) {
    const float e0 = __expf(tw[0]);
    const float e1 = __expf(tw[1]);
    const float s = e0 + e1;
    w[0] = e0 / s;
    w[1] = e1 / s;
  }
}

// ---------------------------------------------------------------------------
extern "C" void kernel_launch(void* const* d_in, const int* in_sizes, int n_in,
                              void* d_out, int out_size, void* d_ws, size_t ws_size,
                              hipStream_t stream) {
  (void)in_sizes; (void)n_in; (void)out_size; (void)ws_size;

  const float* hidden = (const float*)d_in[0];
  struct Track {
    const float *in_w, *conv_w, *conv_b, *x_w, *dtp_w, *dtp_b, *A_log, *D, *mout_w;
  } tr[2];
  for (int t = 0; t < 2; ++t) {
    const int o = 1 + t * 9;
    tr[t].in_w   = (const float*)d_in[o + 0];
    tr[t].conv_w = (const float*)d_in[o + 1];
    tr[t].conv_b = (const float*)d_in[o + 2];
    tr[t].x_w    = (const float*)d_in[o + 3];
    tr[t].dtp_w  = (const float*)d_in[o + 4];
    tr[t].dtp_b  = (const float*)d_in[o + 5];
    tr[t].A_log  = (const float*)d_in[o + 6];
    tr[t].D      = (const float*)d_in[o + 7];
    tr[t].mout_w = (const float*)d_in[o + 8];
  }
  const float* norm_w  = (const float*)d_in[19];
  const float* norm_b  = (const float*)d_in[20];
  const float* track_w = (const float*)d_in[21];
  const float* proj_w  = (const float*)d_in[22];
  const float* proj_b  = (const float*)d_in[23];
  float* outp = (float*)d_out;

  // Workspace carve-up (deterministic, 256B aligned)
  char* ws = (char*)d_ws;
  size_t off = 0;
  auto carve = [&](size_t bytes) -> char* {
    char* p = ws + off;
    off += (bytes + 255) & ~(size_t)255;
    return p;
  };
  _Float16* hn16   = (_Float16*)carve((size_t)BLrows * DM * 2);
  _Float16* w16    = (_Float16*)carve((size_t)2 * DI * DM * 2);  // reused
  float*    xz     = (float*)   carve((size_t)BLrows * 2 * DI * 4);
  float*    xs32   = (float*)   carve((size_t)BLrows * DI * 4);
  _Float16* xs16   = (_Float16*)carve((size_t)BLrows * DI * 2);
  float*    xdbl32 = (float*)   carve((size_t)BLrows * XD * 4);
  _Float16* xdbl16 = (_Float16*)carve((size_t)BLrows * XD * 2);
  float*    dt32   = (float*)   carve((size_t)BLrows * DI * 4);
  float*    y32    = (float*)   carve((size_t)BLrows * DI * 4);
  _Float16* y16    = (_Float16*)carve((size_t)BLrows * DI * 2);
  float*    comb32 = (float*)   carve((size_t)BLrows * DM * 4);
  _Float16* comb16 = (_Float16*)carve((size_t)BLrows * DM * 2);
  float*    hchunk = (float*)   carve((size_t)Bb * NCH * DI * NST * 4);
  float*    hinit  = (float*)   carve((size_t)Bb * NCH * DI * NST * 4);
  float*    sumdt  = (float*)   carve((size_t)Bb * NCH * DI * 4);
  float*    wbuf   = (float*)   carve(256);

  auto cvt = [&](const float* src, _Float16* dst, int n) {
    cvt_f32_f16_kernel<<<(n + 255) / 256, 256, 0, stream>>>(src, dst, n);
  };
  auto gemm = [&](const _Float16* A, int lda, const _Float16* Wm, int ldw,
                  float* C, int ldc, int M, int N, int K,
                  const float* bias, const float* scale, int mode) {
    dim3 grid((N + BN - 1) / BN, M / BM);
    gemm_wmma_kernel<<<grid, 256, 0, stream>>>(A, lda, Wm, ldw, C, ldc, M, N, K,
                                               bias, scale, mode);
  };

  // residual output (second element of the returned tuple)
  (void)hipMemcpyAsync(outp + (size_t)BLrows * DM, hidden,
                       (size_t)BLrows * DM * sizeof(float),
                       hipMemcpyDeviceToDevice, stream);

  softmax2_kernel<<<1, 32, 0, stream>>>(track_w, wbuf);
  layernorm_kernel<<<BLrows, 256, 0, stream>>>(hidden, norm_w, norm_b, hn16);

  const int scan_blocks = Bb * 8 * NCH;   // 256 workgroups
  for (int t = 0; t < 2; ++t) {
    // xz = hn @ in_w.T      (4096 x 4096 x 1024)
    cvt(tr[t].in_w, w16, 2 * DI * DM);
    gemm(hn16, DM, w16, DM, xz, 2 * DI, BLrows, 2 * DI, DM,
         nullptr, nullptr, EPI_STORE);

    // depthwise causal conv + SiLU
    conv_silu_kernel<<<(Bb * Lseq * DI) / 256, 256, 0, stream>>>(
        xz, tr[t].conv_w, tr[t].conv_b, xs32, xs16);

    // x_dbl = xs @ x_w.T    (4096 x 96 x 2048)
    cvt(tr[t].x_w, w16, XD * DI);
    gemm(xs16, DI, w16, DI, xdbl32, XD, BLrows, XD, DI,
         nullptr, nullptr, EPI_STORE);
    cvt(xdbl32, xdbl16, BLrows * XD);

    // dt = softplus(x_dbl[:, :64] @ dtp_w.T + dtp_b)   (4096 x 2048 x 64)
    cvt(tr[t].dtp_w, w16, Rr * DI);
    gemm(xdbl16, XD, w16, Rr, dt32, DI, BLrows, DI, Rr,
         tr[t].dtp_b, nullptr, EPI_SOFTPLUS);

    // chunked selective scan (3 phases) + fused epilogue
    scan_phase1_kernel<<<scan_blocks, 256, 0, stream>>>(
        dt32, xs32, xdbl32, tr[t].A_log, hchunk, sumdt);
    scan_phase2_kernel<<<(Bb * DI * NST) / 256, 256, 0, stream>>>(
        sumdt, hchunk, tr[t].A_log, hinit);
    scan_phase3_kernel<<<scan_blocks, 256, 0, stream>>>(
        dt32, xdbl32, xs32, xz, tr[t].A_log, tr[t].D, hinit, y32, y16);

    // combined {=, +=} w_t * (y @ mout_w.T)   (4096 x 1024 x 2048)
    cvt(tr[t].mout_w, w16, DM * DI);
    gemm(y16, DI, w16, DI, comb32, DM, BLrows, DM, DI,
         nullptr, wbuf + t, (t == 0) ? EPI_SCALE_STORE : EPI_SCALE_ADD);
  }

  // out = combined @ proj_w.T + proj_b   (4096 x 1024 x 1024)
  cvt(comb32, comb16, BLrows * DM);
  cvt(proj_w, w16, DM * DM);
  gemm(comb16, DM, w16, DM, outp, DM, BLrows, DM, DM,
       proj_b, nullptr, EPI_STORE);
}
